// DSTDGC_20057497272946
// MI455X (gfx1250) — matrix operand order repaired and checked by reading, not compiled, not run on hardware
//
#include <hip/hip_runtime.h>
#include <hip/hip_bf16.h>
#include <math.h>

// ---- dims (fixed by reference) ----
#define NN 64
#define TD 64
#define VD 64
#define CD 64
#define KD 128   // RED*T
#define OD 64
#define XMP 132  // padded k-stride for k-major LDS tiles (16B aligned, bank-spread)

typedef __attribute__((ext_vector_type(16))) __bf16        v16bf;
typedef __attribute__((ext_vector_type(16))) unsigned short v16us;
typedef __attribute__((ext_vector_type(8)))  float          v8f;

#if __has_builtin(__builtin_amdgcn_tanhf)
#define TANHF(x) __builtin_amdgcn_tanhf(x)
#else
#define TANHF(x) tanhf(x)
#endif

__device__ __forceinline__ unsigned short f2bf(float f) {
    return __builtin_bit_cast(unsigned short, (__bf16)f);
}

__device__ __forceinline__ v8f wmma_bf16(v16us a, v16us b, v8f c) {
    // 8 args: (neg_a, A, neg_b, B, c_mod, C, reuse_a, reuse_b)
    return __builtin_amdgcn_wmma_f32_16x16x32_bf16(
        false, __builtin_bit_cast(v16bf, a),
        false, __builtin_bit_cast(v16bf, b),
        (short)0, c, false, false);
}

// A-matrix element->K mapping for 16-bit A 16x32 (contiguous, matches B packing):
// lane<16: e0..15 -> K0..15 ; lane>=16: +16
__device__ __forceinline__ int a_klocal(int e, int hi16) {
    return e + hi16 * 16;
}

// ------------------------------------------------------------------
// Kernel 1: xm1/xm2 = 1x1 conv to RED=2 channels, layout (n, r*T+t, v)
// ------------------------------------------------------------------
__global__ __launch_bounds__(256) void k_xm(
    const float* __restrict__ x,
    const float* __restrict__ w_m1, const float* __restrict__ b_m1,
    const float* __restrict__ w_m2, const float* __restrict__ b_m2,
    float* __restrict__ xm1, float* __restrict__ xm2)
{
    __shared__ float w1S[2 * CD];
    __shared__ float w2S[2 * CD];
    int tid = threadIdx.x;
    if (tid < 128)      w1S[tid]       = w_m1[tid];
    else                w2S[tid - 128] = w_m2[tid - 128];
    __syncthreads();

    int g = blockIdx.x * 256 + tid;            // 0 .. N*T*V-1
    int v = g & 63, t = (g >> 6) & 63, n = g >> 12;

    const float* xp = x + (size_t)g * CD;      // x[n][t][v][*] contiguous
    float a0 = b_m1[0], a1 = b_m1[1];
    float c0 = b_m2[0], c1 = b_m2[1];
    #pragma unroll 8
    for (int c = 0; c < CD; ++c) {
        float xv = xp[c];
        a0 += xv * w1S[c];  a1 += xv * w1S[CD + c];
        c0 += xv * w2S[c];  c1 += xv * w2S[CD + c];
    }
    int base = n * KD * VD;
    xm1[base + (t)      * VD + v] = a0;        // k = 0*T + t
    xm1[base + (64 + t) * VD + v] = a1;        // k = 1*T + t
    xm2[base + (t)      * VD + v] = c0;
    xm2[base + (64 + t) * VD + v] = c1;
}

// ------------------------------------------------------------------
// Kernel 2: adj[n,t,i,j] = alpha*(sum_k w_rm[t,k]*tanh(xm1[k,i]-xm2[k,j]) + b_rm[t]) + A[t,i,j]
// One workgroup = (n, 16-row i-tile). GEMM M=t(64), K=128, N=ij(1024).
// w_rm pre-packed in LDS in exact WMMA A-fragment layout; xm tiles k-major.
// Writes adj into d_out (same 64^4 shape as the final output).
// ------------------------------------------------------------------
__global__ __launch_bounds__(256) void k_adj(
    const float* __restrict__ xm1, const float* __restrict__ xm2,
    const float* __restrict__ w_rm, const float* __restrict__ b_rm,
    const float* __restrict__ A, const int* __restrict__ alpha_p,
    float* __restrict__ adj_out)
{
    __shared__ float xm1S[16 * XMP];                 // ~8 KB : xm1[i_local][k]
    __shared__ float xm2S[VD * XMP];                 // ~33 KB : xm2[j][k]
    __shared__ unsigned short wrmPk[4 * 4 * 32 * 16]; // 16 KB : A frags [kb][tt][lane][e]

    int tid = threadIdx.x;
    int n   = blockIdx.x;             // 0..63
    int i0  = blockIdx.y * 16;        // i-tile origin

    const float* xm1g = xm1 + (size_t)n * KD * VD;
    const float* xm2g = xm2 + (size_t)n * KD * VD;
    for (int idx = tid; idx < 16 * KD; idx += 256) {   // [k][il] order: coalesced-ish
        int il = idx & 15, k = idx >> 4;
        xm1S[il * XMP + k] = xm1g[k * VD + i0 + il];
    }
    for (int idx = tid; idx < KD * VD; idx += 256) {   // coalesced global read
        int v = idx & 63, k = idx >> 6;
        xm2S[v * XMP + k] = xm2g[idx];
    }
    // pre-pack w_rm into per-lane A fragments (each element converted once)
    for (int idx = tid; idx < 8192; idx += 256) {
        int e = idx & 15, ln = (idx >> 4) & 31, tt = (idx >> 9) & 3, kb = idx >> 11;
        int hi = ln >> 4, l = ln & 15;
        int kl = a_klocal(e, hi);
        wrmPk[idx] = f2bf(w_rm[(tt * 16 + l) * KD + kb * 32 + kl]);
    }
    __syncthreads();

    float alpha = (float)(*alpha_p);
    int lane = tid & 31, w = tid >> 5;
    int hi16 = (lane >> 4) & 1, l15 = lane & 15;

    for (int q = 0; q < 8; ++q) {                 // 8 column-tiles per wave
        int ct = w * 8 + q;                       // 16 cols: fixed i_local, j-range
        int il = ct >> 2;
        int j0 = (ct & 3) * 16;

        const float* x1r = &xm1S[il * XMP];
        const float* x2r = &xm2S[(j0 + l15) * XMP];

        v8f acc[4];
        #pragma unroll
        for (int tt = 0; tt < 4; ++tt) acc[tt] = (v8f){0,0,0,0,0,0,0,0};

        #pragma unroll
        for (int kb = 0; kb < 4; ++kb) {          // K = 4 x 32
            int kbase = kb * 32 + hi16 * 16;      // 16 k-consecutive -> b128 LDS loads
            v16us bf;                             // B: tanh(xm1_i - xm2_j), bf16
            #pragma unroll
            for (int e = 0; e < 16; ++e) {
                float d = x1r[kbase + e] - x2r[kbase + e];
                bf[e] = f2bf(TANHF(d));
            }
            #pragma unroll
            for (int tt = 0; tt < 4; ++tt) {      // 4 t-tiles share this B
                v16us af = *(const v16us*)&wrmPk[((kb * 4 + tt) * 32 + lane) * 16];
                acc[tt] = wmma_bf16(af, bf, acc[tt]);
            }
        }

        int i = i0 + il, j = j0 + l15;
        #pragma unroll
        for (int tt = 0; tt < 4; ++tt) {
            #pragma unroll
            for (int r = 0; r < 8; ++r) {
                int t = tt * 16 + r + hi16 * 8;
                float vv = (acc[tt][r] + b_rm[t]) * alpha + A[(t * VD + i) * VD + j];
                adj_out[(((size_t)(n * TD + t) * VD) + i) * VD + j] = vv;
            }
        }
    }
}

// ------------------------------------------------------------------
// Kernel 3: per (n,t):  xf = x[n,t] @ w_f^T + b_f  (WMMA),
//                       out = adj[n,t] @ xf        (WMMA), in-place over d_out.
// adj tile fully staged in LDS before overwrite -> safe. w_f and xf are
// pre-packed into bf16 B-fragment layout so the GEMM loop is pure ds+wmma.
// ------------------------------------------------------------------
__global__ __launch_bounds__(256) void k_out(
    const float* __restrict__ x, const float* __restrict__ w_f,
    const float* __restrict__ b_f, float* __restrict__ out)
{
    __shared__ float adjS[VD * VD];                   // 16 KB
    __shared__ float xfS[VD * OD];                    // 16 KB
    __shared__ unsigned short wfPk[4 * 2 * 32 * 16];  // 8 KB : B frags [ot][kb][lane][e]
    __shared__ unsigned short xfPk[4 * 2 * 32 * 16];  // 8 KB : B frags [ot][kb][lane][e]

    int tid = threadIdx.x;
    size_t blk = blockIdx.x;          // n*64 + t
    const float* adjg = out + blk * 4096;
    const float* xg   = x   + blk * 4096;
    for (int idx = tid; idx < 4096; idx += 256) adjS[idx] = adjg[idx];
    // pack w_f^T into B fragments: B[c,o] = w_f[o][c]
    for (int idx = tid; idx < 4096; idx += 256) {
        int e = idx & 15, ln = (idx >> 4) & 31, kb = (idx >> 9) & 1, ot = idx >> 10;
        int hi = ln >> 4, l = ln & 15;
        int c = kb * 32 + e + hi * 16;
        wfPk[idx] = f2bf(w_f[(ot * 16 + l) * CD + c]);
    }
    __syncthreads();

    int lane = tid & 31, w = tid >> 5;
    int hi16 = (lane >> 4) & 1, l15 = lane & 15;

    // ---- GEMM1: xf[j,o] = sum_c x[j,c] * w_f[o,c] + b_f[o] ----
    int jt = w >> 1;                          // both tiles of this wave share jt
    v16us ax[2];
    #pragma unroll
    for (int kb = 0; kb < 2; ++kb) {          // A: x[j,c] straight from global
        #pragma unroll
        for (int e = 0; e < 16; ++e) {
            int kl = a_klocal(e, hi16);
            ax[kb][e] = f2bf(xg[(jt * 16 + l15) * CD + kb * 32 + kl]);
        }
    }
    #pragma unroll
    for (int p = 0; p < 2; ++p) {
        int ot = (w * 2 + p) & 3;
        v8f acc = (v8f){0,0,0,0,0,0,0,0};
        #pragma unroll
        for (int kb = 0; kb < 2; ++kb) {
            v16us bfv = *(const v16us*)&wfPk[((ot * 2 + kb) * 32 + lane) * 16];
            acc = wmma_bf16(ax[kb], bfv, acc);
        }
        float bias = b_f[ot * 16 + l15];
        #pragma unroll
        for (int r = 0; r < 8; ++r) {
            int j = jt * 16 + r + hi16 * 8;
            xfS[j * OD + ot * 16 + l15] = acc[r] + bias;
        }
    }
    __syncthreads();

    // pack xf into B fragments for GEMM2: B[j,o]
    for (int idx = tid; idx < 4096; idx += 256) {
        int e = idx & 15, ln = (idx >> 4) & 31, kb = (idx >> 9) & 1, ot = idx >> 10;
        int hi = ln >> 4, l = ln & 15;
        int j = kb * 32 + e + hi * 16;
        xfPk[idx] = f2bf(xfS[j * OD + ot * 16 + l]);
    }
    __syncthreads();

    // ---- GEMM2: out[i,o] = sum_j adj[i,j] * xf[j,o] ----
    int it = w >> 1;
    v16us aa[2];
    #pragma unroll
    for (int kb = 0; kb < 2; ++kb) {          // A: adj[i,j] from LDS (b128 runs)
        #pragma unroll
        for (int e = 0; e < 16; ++e) {
            int kl = a_klocal(e, hi16);
            aa[kb][e] = f2bf(adjS[(it * 16 + l15) * VD + kb * 32 + kl]);
        }
    }
    float* og = out + blk * 4096;
    #pragma unroll
    for (int p = 0; p < 2; ++p) {
        int ot = (w * 2 + p) & 3;
        v8f acc = (v8f){0,0,0,0,0,0,0,0};
        #pragma unroll
        for (int kb = 0; kb < 2; ++kb) {
            v16us bfv = *(const v16us*)&xfPk[((ot * 2 + kb) * 32 + lane) * 16];
            acc = wmma_bf16(aa[kb], bfv, acc);
        }
        #pragma unroll
        for (int r = 0; r < 8; ++r) {
            int i = it * 16 + r + hi16 * 8;
            og[i * OD + ot * 16 + l15] = acc[r];
        }
    }
}

// ------------------------------------------------------------------
extern "C" void kernel_launch(void* const* d_in, const int* in_sizes, int n_in,
                              void* d_out, int out_size, void* d_ws, size_t ws_size,
                              hipStream_t stream) {
    (void)in_sizes; (void)n_in; (void)out_size; (void)ws_size;
    const float* x     = (const float*)d_in[0];
    const float* A     = (const float*)d_in[1];
    const float* w_m1  = (const float*)d_in[2];
    const float* b_m1  = (const float*)d_in[3];
    const float* w_m2  = (const float*)d_in[4];
    const float* b_m2  = (const float*)d_in[5];
    const float* w_rm  = (const float*)d_in[6];
    const float* b_rm  = (const float*)d_in[7];
    const float* w_f   = (const float*)d_in[8];
    const float* b_f   = (const float*)d_in[9];
    const int*   alpha = (const int*)d_in[10];
    float* out = (float*)d_out;

    float* xm1 = (float*)d_ws;                       // N*K*V = 524288 floats
    float* xm2 = xm1 + (size_t)NN * KD * VD;         // + 524288 floats (4 MB total)

    k_xm<<<(NN * TD * VD) / 256, 256, 0, stream>>>(x, w_m1, b_m1, w_m2, b_m2, xm1, xm2);

    dim3 gAdj(NN, VD / 16);                          // (n, i-tile)
    k_adj<<<gAdj, 256, 0, stream>>>(xm1, xm2, w_rm, b_rm, A, alpha, out);

    k_out<<<NN * TD, 256, 0, stream>>>(x, w_f, b_f, out);
}
